// BidirectionalCrossAttention_23648089932299
// MI455X (gfx1250) — matrix operand order repaired and checked by reading, not compiled, not run on hardware
//
#include <hip/hip_runtime.h>

// ---------------------------------------------------------------------------
// Bidirectional cross attention for MI455X (gfx1250, wave32, WMMA + TDM).
// B=2, N=M=2048, DIM=INNER=512, HEADS=8, DIM_HEAD=64.
//
// - bf16 WMMA (v_wmma_f32_16x16x32_bf16) for every matmul, fp32 accumulate.
// - sim materialized bf16 (134 MB < 192 MB L2 -> the 4 re-read passes are
//   L2-resident instead of HBM traffic).
// - k_sim tile staging via Tensor Data Mover (tensor_load_to_lds, 6-arg form)
//   with D# LDS-padding reproducing the bank-conflict-avoiding stride.
// - k_gemm512 A-tile staging via global_load_async_to_lds_b128 (ASYNCcnt).
// ---------------------------------------------------------------------------

typedef unsigned short u16;
typedef __bf16 bf16_t;
typedef bf16_t v16bf __attribute__((ext_vector_type(16)));
typedef float  v8f   __attribute__((ext_vector_type(8)));
typedef unsigned int u32x4 __attribute__((ext_vector_type(4)));
typedef int          i32x4 __attribute__((ext_vector_type(4)));
typedef int          i32x8 __attribute__((ext_vector_type(8)));

#define BATCH 2
#define NSEQ  2048
#define DIM   512
#define HEADS 8
#define DHEAD 64
#define BHTOT (BATCH*HEADS)

#if defined(__has_builtin)
#if __has_builtin(__builtin_amdgcn_tensor_load_to_lds)
#define HAVE_TDM 1
#endif
#endif

__device__ __forceinline__ float bf2f(u16 h) {
    union { unsigned u; float f; } v; v.u = ((unsigned)h) << 16; return v.f;
}
__device__ __forceinline__ u16 f2bf(float f) {
    union { float f; unsigned u; } v; v.f = f;
    unsigned r = v.u + 0x7fffu + ((v.u >> 16) & 1u);
    return (u16)(r >> 16);
}
__device__ __forceinline__ v8f vzero() {
    v8f z;
#pragma unroll
    for (int i = 0; i < 8; ++i) z[i] = 0.0f;
    return z;
}
__device__ __forceinline__ unsigned rfl(unsigned x) {
    return (unsigned)__builtin_amdgcn_readfirstlane((int)x);
}
// Low 32 bits of an aperture-based generic LDS pointer == LDS byte offset
// (see ISA 10.2).
__device__ __forceinline__ unsigned lds_off(const void* p) {
    return (unsigned)(unsigned long long)(uintptr_t)p;
}

union Frag {                 // one 16x32 bf16 WMMA operand (32 bytes / lane)
    uint4 q[2];
    u16   s[16];
    v16bf v;
};
union Q16 { uint4 q; u16 s[8]; };

__device__ __forceinline__ v8f wmma_bf16(const Frag& a, const Frag& b, v8f c) {
    return __builtin_amdgcn_wmma_f32_16x16x32_bf16(
        /*neg_a=*/false, a.v, /*neg_b=*/false, b.v,
        /*c_mod=*/(short)0, c, /*reuse_a=*/false, /*reuse_b=*/false);
}

// ---- async global->LDS copy (per-lane, tracked by ASYNCcnt) ---------------
__device__ __forceinline__ void async_copy_b128(unsigned lds_addr, const void* gaddr) {
    unsigned long long ga = (unsigned long long)(uintptr_t)gaddr;
    asm volatile("global_load_async_to_lds_b128 %0, %1, off"
                 :: "v"(lds_addr), "v"(ga) : "memory");
}
__device__ __forceinline__ void wait_asynccnt0() {
    asm volatile("s_wait_asynccnt 0x0" ::: "memory");
}
__device__ __forceinline__ void wait_tensorcnt0() {
#if defined(__has_builtin) && __has_builtin(__builtin_amdgcn_s_wait_tensorcnt)
    __builtin_amdgcn_s_wait_tensorcnt((short)0);
#else
    asm volatile("s_wait_tensorcnt 0x0" ::: "memory");
#endif
}

#ifdef HAVE_TDM
// TDM 2D tile load: tile_d0 x tile_d1 elements of 2 bytes, row stride
// stride0 (elements), optional LDS padding (pad_interval/pad_amount encodings
// per D# group1). Descriptor words forced uniform via readfirstlane.
// 6-arg builtin form: (g0 u32x4, g1 i32x8, g2 i32x4, g3 i32x4, i32x8, cpol).
__device__ __forceinline__ void tdm_load_2d(unsigned lds_addr, const void* gptr,
                                            unsigned tensor_d0, unsigned tensor_d1,
                                            unsigned tile_d0, unsigned tile_d1,
                                            unsigned long long stride0,
                                            unsigned pad_en, unsigned pad_int,
                                            unsigned pad_amt) {
    unsigned long long ga = (unsigned long long)(uintptr_t)gptr;
    u32x4 g0;
    g0[0] = rfl(1u);                                    // count=1, user desc
    g0[1] = rfl(lds_addr);                              // lds_addr
    g0[2] = rfl((unsigned)ga);                          // global_addr[31:0]
    g0[3] = rfl((unsigned)((ga >> 32) & 0x1FFFFFFull)   // global_addr[56:32]
                | (2u << 30));                          // type=2 ("image")
    i32x8 g1;
    g1[0] = (int)rfl((1u << 16)                         // data_size=2B
                     | (pad_en << 20) | (pad_int << 22) | (pad_amt << 25));
    g1[1] = (int)rfl(tensor_d0 << 16);                  // tensor_dim0[15:0]@48
    g1[2] = (int)rfl((tensor_d0 >> 16) | (tensor_d1 << 16));
    g1[3] = (int)rfl((tensor_d1 >> 16) | (tile_d0 << 16));
    g1[4] = (int)rfl(tile_d1 & 0xFFFFu);                // tile_dim1; dim2=0
    g1[5] = (int)rfl((unsigned)(stride0 & 0xFFFFFFFFull));
    g1[6] = (int)rfl((unsigned)((stride0 >> 32) & 0xFFFFull));
    g1[7] = 0;
    i32x4 z4; z4[0] = z4[1] = z4[2] = z4[3] = 0;
    i32x8 z8;
#pragma unroll
    for (int i = 0; i < 8; ++i) z8[i] = 0;
    __builtin_amdgcn_tensor_load_to_lds(g0, g1, z4, z4, z8, 0);
}
#endif

// ---------------------------------------------------------------- fp32->bf16
__global__ void k_f2bf(const float* __restrict__ src, u16* __restrict__ dst, int n) {
    int i = blockIdx.x * blockDim.x + threadIdx.x;
    if (i < n) dst[i] = f2bf(src[i]);
}

// ------------------------------------------------- 512-col, K=512 WMMA GEMM
// mode 0: store bf16 head-split [b,h,n,64]      (projections)
// mode 1: store fp32 flat [row,512] + bias      (output projections)
__global__ __launch_bounds__(256)
void k_gemm512(const u16* __restrict__ A, const u16* __restrict__ W,
               void* __restrict__ Cout, const float* __restrict__ bias, int mode) {
    __shared__ u16 sA[128][48];    // [m][k]  stride 96B (16B aligned frags)
    __shared__ u16 sBt[128][48];   // [n][k]  (W tile transposed)
    const int tid  = threadIdx.x;
    const int lane = tid & 31, wave = tid >> 5;
    const int wm = wave & 3, wn = wave >> 2;          // 4x2 wave grid
    const int rowBase = blockIdx.y * 128;
    const int colBase = blockIdx.x * 128;
    const int l15 = lane & 15;

    v8f acc[2][4];
#pragma unroll
    for (int i = 0; i < 2; ++i)
#pragma unroll
        for (int j = 0; j < 4; ++j) acc[i][j] = vzero();

    for (int k0 = 0; k0 < DIM; k0 += 32) {
#pragma unroll
        for (int i = 0; i < 4; ++i) {                  // A tile 128x32 (async)
            int idx = tid + i * 256;
            int r = idx >> 2, cg = idx & 3;
            async_copy_b128(lds_off(&sA[r][cg * 8]),
                            A + (size_t)(rowBase + r) * DIM + k0 + cg * 8);
        }
#pragma unroll
        for (int i = 0; i < 2; ++i) {                  // W tile 32x128, transposed
            int idx = tid + i * 256;
            int kk = idx >> 4, cg = idx & 15;
            Q16 d; d.q = *(const uint4*)(W + (size_t)(k0 + kk) * DIM + colBase + cg * 8);
#pragma unroll
            for (int e = 0; e < 8; ++e) sBt[cg * 8 + e][kk] = d.s[e];
        }
        wait_asynccnt0();
        __syncthreads();
        const int kbA = (lane < 16) ? 0 : 8;
        const int kbB = (lane < 16) ? 0 : 8;
        Frag bfr[4];
#pragma unroll
        for (int ni = 0; ni < 4; ++ni) {
            int n = wn * 64 + ni * 16 + l15;
            bfr[ni].q[0] = *(const uint4*)&sBt[n][kbB];
            bfr[ni].q[1] = *(const uint4*)&sBt[n][kbB + 16];
        }
#pragma unroll
        for (int mi = 0; mi < 2; ++mi) {
            int m = wm * 32 + mi * 16 + l15;
            Frag a;
            a.q[0] = *(const uint4*)&sA[m][kbA];
            a.q[1] = *(const uint4*)&sA[m][kbA + 16];
#pragma unroll
            for (int ni = 0; ni < 4; ++ni)
                acc[mi][ni] = wmma_bf16(a, bfr[ni], acc[mi][ni]);
        }
        __syncthreads();
    }

    const int mb = (lane < 16) ? 0 : 8;
#pragma unroll
    for (int mi = 0; mi < 2; ++mi)
#pragma unroll
        for (int ni = 0; ni < 4; ++ni)
#pragma unroll
            for (int r = 0; r < 8; ++r) {
                int grow = rowBase + wm * 32 + mi * 16 + mb + r;
                int gcol = colBase + wn * 64 + ni * 16 + l15;
                float val = acc[mi][ni][r];
                if (mode == 0) {
                    int bb = grow >> 11, n = grow & 2047;
                    int h = gcol >> 6, d = gcol & 63;
                    ((u16*)Cout)[((size_t)(bb * HEADS + h) * NSEQ + n) * DHEAD + d] = f2bf(val);
                } else {
                    ((float*)Cout)[(size_t)grow * DIM + gcol] = val + bias[gcol];
                }
            }
}

// --------------------------------------------- S = scale * QK @ CQK^T (bf16)
__global__ __launch_bounds__(256)
void k_sim(const u16* __restrict__ qk, const u16* __restrict__ cqk,
           u16* __restrict__ sim) {
    __shared__ u16 sQ[128][80];    // [n_local][d]  stride 160B (TDM-padded)
    __shared__ u16 sC[128][80];    // [m_local][d]
    const int tid = threadIdx.x, lane = tid & 31, wave = tid >> 5;
    const int wm = wave & 3, wn = wave >> 2;
    const int bh = blockIdx.z;
    const int nBase = blockIdx.y * 128;
    const int mBase = blockIdx.x * 128;
    const int l15 = lane & 15;
    const size_t hOff = (size_t)bh * NSEQ * DHEAD;

#ifdef HAVE_TDM
    // One wave drives the DMA: 128x64 bf16 tile each, row = 128B of data then
    // 32B LDS pad -> pad_interval=4 (32 DWORDs), pad_amount=7 (8 DWORDs).
    if (wave == 0) {
        tdm_load_2d(lds_off(&sQ[0][0]), qk  + hOff + (size_t)nBase * DHEAD,
                    DHEAD, NSEQ, DHEAD, 128, DHEAD, 1u, 4u, 7u);
        tdm_load_2d(lds_off(&sC[0][0]), cqk + hOff + (size_t)mBase * DHEAD,
                    DHEAD, NSEQ, DHEAD, 128, DHEAD, 1u, 4u, 7u);
        wait_tensorcnt0();
    }
    __syncthreads();
#else
#pragma unroll
    for (int i = 0; i < 4; ++i) {
        int idx = tid + i * 256;
        int r = idx >> 3, cg = idx & 7;
        *(uint4*)&sQ[r][cg * 8] = *(const uint4*)(qk  + hOff + (size_t)(nBase + r) * DHEAD + cg * 8);
        *(uint4*)&sC[r][cg * 8] = *(const uint4*)(cqk + hOff + (size_t)(mBase + r) * DHEAD + cg * 8);
    }
    __syncthreads();
#endif

    v8f acc[2][4];
#pragma unroll
    for (int i = 0; i < 2; ++i)
#pragma unroll
        for (int j = 0; j < 4; ++j) acc[i][j] = vzero();

#pragma unroll
    for (int ks = 0; ks < 64; ks += 32) {
        int kbA = ks + ((lane < 16) ? 0 : 8);
        int kbB = ks + ((lane < 16) ? 0 : 8);
        Frag bfr[4];
#pragma unroll
        for (int ni = 0; ni < 4; ++ni) {
            int n = wn * 64 + ni * 16 + l15;
            bfr[ni].q[0] = *(const uint4*)&sC[n][kbB];
            bfr[ni].q[1] = *(const uint4*)&sC[n][kbB + 16];
        }
#pragma unroll
        for (int mi = 0; mi < 2; ++mi) {
            int m = wm * 32 + mi * 16 + l15;
            Frag a;
            a.q[0] = *(const uint4*)&sQ[m][kbA];
            a.q[1] = *(const uint4*)&sQ[m][kbA + 16];
#pragma unroll
            for (int ni = 0; ni < 4; ++ni)
                acc[mi][ni] = wmma_bf16(a, bfr[ni], acc[mi][ni]);
        }
    }
    const float scale = 0.125f;   // 1/sqrt(64)
    const int mb = (lane < 16) ? 0 : 8;
#pragma unroll
    for (int mi = 0; mi < 2; ++mi)
#pragma unroll
        for (int ni = 0; ni < 4; ++ni)
#pragma unroll
            for (int r = 0; r < 8; ++r) {
                int gi = nBase + wm * 32 + mi * 16 + mb + r;
                int gj = mBase + wn * 64 + ni * 16 + l15;
                sim[((size_t)bh * NSEQ + gi) * NSEQ + gj] = f2bf(acc[mi][ni][r] * scale);
            }
}

// ---------------------------------------------------------- row softmax stats
__global__ __launch_bounds__(256)
void k_rowstats(const u16* __restrict__ sim, float* __restrict__ rmax,
                float* __restrict__ rsum) {
    __shared__ float red[256];
    const int tid = threadIdx.x;
    const size_t row = (size_t)blockIdx.y * NSEQ + blockIdx.x;
    const u16* p = sim + row * NSEQ;
    Q16 d; d.q = *(const uint4*)(p + tid * 8);
    float v[8]; float m = -1e30f;
#pragma unroll
    for (int e = 0; e < 8; ++e) { v[e] = bf2f(d.s[e]); m = fmaxf(m, v[e]); }
    red[tid] = m; __syncthreads();
    for (int s = 128; s > 0; s >>= 1) {
        if (tid < s) red[tid] = fmaxf(red[tid], red[tid + s]);
        __syncthreads();
    }
    m = red[0]; __syncthreads();
    float sum = 0.f;
#pragma unroll
    for (int e = 0; e < 8; ++e) sum += __expf(v[e] - m);
    red[tid] = sum; __syncthreads();
    for (int s = 128; s > 0; s >>= 1) {
        if (tid < s) red[tid] += red[tid + s];
        __syncthreads();
    }
    if (tid == 0) { rmax[row] = m; rsum[row] = red[0]; }
}

// ------------------------------------------------------- column softmax stats
__global__ __launch_bounds__(256)
void k_colstats(const u16* __restrict__ sim, float* __restrict__ cmax,
                float* __restrict__ csum) {
    __shared__ float mArr[256], sArr[256];
    const int tid = threadIdx.x;
    const int bh = blockIdx.y;
    const int col = blockIdx.x * 64 + (tid & 63);
    const int rg = tid >> 6;
    const u16* base = sim + (size_t)bh * NSEQ * NSEQ;
    float m = -1e30f, s = 0.f;
    for (int i = rg; i < NSEQ; i += 4) {      // online max/sum, coalesced rows
        float x = bf2f(base[(size_t)i * NSEQ + col]);
        if (x > m) { s = s * __expf(m - x) + 1.f; m = x; }
        else        s += __expf(x - m);
    }
    mArr[tid] = m; sArr[tid] = s; __syncthreads();
    if (tid < 64) {
        for (int g = 1; g < 4; ++g) {
            float m2 = mArr[tid + g * 64], s2 = sArr[tid + g * 64];
            if (m2 > m) { s = s * __expf(m - m2) + s2; m = m2; }
            else          s += s2 * __expf(m2 - m);
        }
        cmax[(size_t)bh * NSEQ + col] = m;
        csum[(size_t)bh * NSEQ + col] = s;
    }
}

// ------------------------ out[b,h,i,:] = (exp(S-rowmax) @ cv) / rowsum  (bf16)
__global__ __launch_bounds__(256)
void k_attn_row(const u16* __restrict__ sim, const u16* __restrict__ cv,
                const float* __restrict__ rmax, const float* __restrict__ rsum,
                u16* __restrict__ outh) {
    __shared__ u16 sP[128][48];
    __shared__ u16 sVt[64][48];
    __shared__ float sRmax[128];
    const int tid = threadIdx.x, lane = tid & 31, wave = tid >> 5;
    const int bh = blockIdx.y, b = bh >> 3, h = bh & 7;
    const int iBase = blockIdx.x * 128;
    const int l15 = lane & 15;
    const u16* simB = sim + (size_t)bh * NSEQ * NSEQ;
    const u16* cvB  = cv  + (size_t)bh * NSEQ * DHEAD;
    if (tid < 128) sRmax[tid] = rmax[(size_t)bh * NSEQ + iBase + tid];
    v8f acc[4];
#pragma unroll
    for (int j = 0; j < 4; ++j) acc[j] = vzero();
    __syncthreads();

    for (int j0 = 0; j0 < NSEQ; j0 += 32) {
        {   // P tile = exp(S - rowmax), 128x32 bf16
            int r = tid >> 1, c = (tid & 1) * 16;
            Frag d, o;
            const u16* src = simB + (size_t)(iBase + r) * NSEQ + j0 + c;
            d.q[0] = *(const uint4*)src; d.q[1] = *(const uint4*)(src + 8);
            float rm = sRmax[r];
#pragma unroll
            for (int e = 0; e < 16; ++e) o.s[e] = f2bf(__expf(bf2f(d.s[e]) - rm));
            *(uint4*)&sP[r][c] = o.q[0];
            *(uint4*)&sP[r][c + 8] = o.q[1];
        }
        {   // cv chunk 32x64 -> transposed [n][k]
            int k = tid >> 3, cg = tid & 7;
            Q16 d; d.q = *(const uint4*)(cvB + (size_t)(j0 + k) * DHEAD + cg * 8);
#pragma unroll
            for (int e = 0; e < 8; ++e) sVt[cg * 8 + e][k] = d.s[e];
        }
        __syncthreads();
        int kbA = (lane < 16) ? 0 : 8, kbB = (lane < 16) ? 0 : 8;
        int m = wave * 16 + l15;
        Frag a;
        a.q[0] = *(const uint4*)&sP[m][kbA];
        a.q[1] = *(const uint4*)&sP[m][kbA + 16];
#pragma unroll
        for (int ni = 0; ni < 4; ++ni) {
            Frag bb;
            int n = ni * 16 + l15;
            bb.q[0] = *(const uint4*)&sVt[n][kbB];
            bb.q[1] = *(const uint4*)&sVt[n][kbB + 16];
            acc[ni] = wmma_bf16(a, bb, acc[ni]);
        }
        __syncthreads();
    }
    const int mb = (lane < 16) ? 0 : 8;
    float rs[8];
#pragma unroll
    for (int r = 0; r < 8; ++r)
        rs[r] = rsum[(size_t)bh * NSEQ + iBase + wave * 16 + mb + r];
#pragma unroll
    for (int ni = 0; ni < 4; ++ni)
#pragma unroll
        for (int r = 0; r < 8; ++r) {
            int grow = iBase + wave * 16 + mb + r;
            int dcol = ni * 16 + l15;
            outh[((size_t)b * NSEQ + grow) * DIM + h * DHEAD + dcol] =
                f2bf(acc[ni][r] / rs[r]);
        }
}

// --------- context_out[b,h,j,:] = (exp(S^T - colmax) @ v) / colsum  (bf16)
__global__ __launch_bounds__(256)
void k_attn_col(const u16* __restrict__ sim, const u16* __restrict__ vmat,
                const float* __restrict__ cmax, const float* __restrict__ csum,
                u16* __restrict__ couth) {
    __shared__ u16 sP[128][48];     // [j_local][i_local]
    __shared__ u16 sVt[64][48];
    __shared__ float sCmax[128];
    const int tid = threadIdx.x, lane = tid & 31, wave = tid >> 5;
    const int bh = blockIdx.y, b = bh >> 3, h = bh & 7;
    const int jBase = blockIdx.x * 128;
    const int l15 = lane & 15;
    const u16* simB = sim  + (size_t)bh * NSEQ * NSEQ;
    const u16* vB   = vmat + (size_t)bh * NSEQ * DHEAD;
    if (tid < 128) sCmax[tid] = cmax[(size_t)bh * NSEQ + jBase + tid];
    v8f acc[4];
#pragma unroll
    for (int j = 0; j < 4; ++j) acc[j] = vzero();
    __syncthreads();

    for (int i0 = 0; i0 < NSEQ; i0 += 32) {
        {   // read S in natural order, write exp(S - colmax) transposed
            int ii = tid >> 3;
            int jc = (tid & 7) * 16;
            Frag d;
            const u16* src = simB + (size_t)(i0 + ii) * NSEQ + jBase + jc;
            d.q[0] = *(const uint4*)src; d.q[1] = *(const uint4*)(src + 8);
#pragma unroll
            for (int e = 0; e < 16; ++e)
                sP[jc + e][ii] = f2bf(__expf(bf2f(d.s[e]) - sCmax[jc + e]));
        }
        {   // v chunk 32x64 -> transposed [n][k]
            int k = tid >> 3, cg = tid & 7;
            Q16 d; d.q = *(const uint4*)(vB + (size_t)(i0 + k) * DHEAD + cg * 8);
#pragma unroll
            for (int e = 0; e < 8; ++e) sVt[cg * 8 + e][k] = d.s[e];
        }
        __syncthreads();
        int kbA = (lane < 16) ? 0 : 8, kbB = (lane < 16) ? 0 : 8;
        int m = wave * 16 + l15;
        Frag a;
        a.q[0] = *(const uint4*)&sP[m][kbA];
        a.q[1] = *(const uint4*)&sP[m][kbA + 16];
#pragma unroll
        for (int ni = 0; ni < 4; ++ni) {
            Frag bb;
            int n = ni * 16 + l15;
            bb.q[0] = *(const uint4*)&sVt[n][kbB];
            bb.q[1] = *(const uint4*)&sVt[n][kbB + 16];
            acc[ni] = wmma_bf16(a, bb, acc[ni]);
        }
        __syncthreads();
    }
    const int mb = (lane < 16) ? 0 : 8;
    float cs[8];
#pragma unroll
    for (int r = 0; r < 8; ++r)
        cs[r] = csum[(size_t)bh * NSEQ + jBase + wave * 16 + mb + r];
#pragma unroll
    for (int ni = 0; ni < 4; ++ni)
#pragma unroll
        for (int r = 0; r < 8; ++r) {
            int grow = jBase + wave * 16 + mb + r;
            int dcol = ni * 16 + l15;
            couth[((size_t)b * NSEQ + grow) * DIM + h * DHEAD + dcol] =
                f2bf(acc[ni][r] / cs[r]);
        }
}

// ---------------------------------------------------------------------------
extern "C" void kernel_launch(void* const* d_in, const int* in_sizes, int n_in,
                              void* d_out, int out_size, void* d_ws, size_t ws_size,
                              hipStream_t stream) {
    (void)in_sizes; (void)n_in; (void)out_size; (void)ws_size;
    const float* x      = (const float*)d_in[0];
    const float* ctx    = (const float*)d_in[1];
    const float* w_qk   = (const float*)d_in[2];
    const float* w_cqk  = (const float*)d_in[3];
    const float* w_v    = (const float*)d_in[4];
    const float* w_cv   = (const float*)d_in[5];
    const float* w_out  = (const float*)d_in[6];
    const float* b_out  = (const float*)d_in[7];
    const float* w_cout = (const float*)d_in[8];
    const float* b_cout = (const float*)d_in[9];

    char* ws = (char*)d_ws;
    size_t off = 0;
    auto alloc = [&](size_t bytes) -> char* {
        char* p = ws + off;
        off += (bytes + 255) & ~(size_t)255;
        return p;
    };
    const size_t NTOK = (size_t)BATCH * NSEQ;   // 4096
    const size_t EA   = NTOK * DIM;             // 2,097,152
    const size_t EW   = (size_t)DIM * DIM;      // 262,144
    const size_t EH   = (size_t)BHTOT * NSEQ * DHEAD;

    u16* xb     = (u16*)alloc(EA * 2);
    u16* cb     = (u16*)alloc(EA * 2);
    u16* wqkb   = (u16*)alloc(EW * 2);
    u16* wcqkb  = (u16*)alloc(EW * 2);
    u16* wvb    = (u16*)alloc(EW * 2);
    u16* wcvb   = (u16*)alloc(EW * 2);
    u16* woutb  = (u16*)alloc(EW * 2);
    u16* wcoutb = (u16*)alloc(EW * 2);
    u16* qkh    = (u16*)alloc(EH * 2);
    u16* vh     = (u16*)alloc(EH * 2);
    u16* cqkh   = (u16*)alloc(EH * 2);
    u16* cvh    = (u16*)alloc(EH * 2);
    u16* simw   = (u16*)alloc((size_t)BHTOT * NSEQ * NSEQ * 2);   // 134 MB, L2-resident
    float* rmaxb = (float*)alloc((size_t)BHTOT * NSEQ * 4);
    float* rsumb = (float*)alloc((size_t)BHTOT * NSEQ * 4);
    float* cmaxb = (float*)alloc((size_t)BHTOT * NSEQ * 4);
    float* csumb = (float*)alloc((size_t)BHTOT * NSEQ * 4);
    u16* outh    = (u16*)alloc(EA * 2);
    u16* couth   = (u16*)alloc(EA * 2);

    // Stage 0: convert to bf16
    k_f2bf<<<dim3((unsigned)((EA + 255) / 256)), 256, 0, stream>>>(x,      xb,     (int)EA);
    k_f2bf<<<dim3((unsigned)((EA + 255) / 256)), 256, 0, stream>>>(ctx,    cb,     (int)EA);
    k_f2bf<<<dim3((unsigned)((EW + 255) / 256)), 256, 0, stream>>>(w_qk,   wqkb,   (int)EW);
    k_f2bf<<<dim3((unsigned)((EW + 255) / 256)), 256, 0, stream>>>(w_cqk,  wcqkb,  (int)EW);
    k_f2bf<<<dim3((unsigned)((EW + 255) / 256)), 256, 0, stream>>>(w_v,    wvb,    (int)EW);
    k_f2bf<<<dim3((unsigned)((EW + 255) / 256)), 256, 0, stream>>>(w_cv,   wcvb,   (int)EW);
    k_f2bf<<<dim3((unsigned)((EW + 255) / 256)), 256, 0, stream>>>(w_out,  woutb,  (int)EW);
    k_f2bf<<<dim3((unsigned)((EW + 255) / 256)), 256, 0, stream>>>(w_cout, wcoutb, (int)EW);

    // Stage 1: projections (bf16, head-split)
    dim3 gGemm(DIM / 128, (unsigned)(NTOK / 128));
    k_gemm512<<<gGemm, 256, 0, stream>>>(xb, wqkb,  qkh,  nullptr, 0);
    k_gemm512<<<gGemm, 256, 0, stream>>>(xb, wvb,   vh,   nullptr, 0);
    k_gemm512<<<gGemm, 256, 0, stream>>>(cb, wcqkb, cqkh, nullptr, 0);
    k_gemm512<<<gGemm, 256, 0, stream>>>(cb, wcvb,  cvh,  nullptr, 0);

    // Stage 2: shared similarity matrix (TDM-staged tiles)
    k_sim<<<dim3(NSEQ / 128, NSEQ / 128, BHTOT), 256, 0, stream>>>(qkh, cqkh, simw);

    // Stage 3: softmax statistics (row and column directions)
    k_rowstats<<<dim3(NSEQ, BHTOT), 256, 0, stream>>>(simw, rmaxb, rsumb);
    k_colstats<<<dim3(NSEQ / 64, BHTOT), 256, 0, stream>>>(simw, cmaxb, csumb);

    // Stage 4: attention applies
    k_attn_row<<<dim3(NSEQ / 128, BHTOT), 256, 0, stream>>>(simw, cvh, rmaxb, rsumb, outh);
    k_attn_col<<<dim3(NSEQ / 128, BHTOT), 256, 0, stream>>>(simw, vh,  cmaxb, csumb, couth);

    // Stage 5: output projections (fp32 + bias) -> d_out, concatenated
    k_gemm512<<<gGemm, 256, 0, stream>>>(outh,  woutb,  (float*)d_out,       b_out,  1);
    k_gemm512<<<gGemm, 256, 0, stream>>>(couth, wcoutb, (float*)d_out + EA,  b_cout, 1);
}
